// ModulatedConv2D_44555990729382
// MI455X (gfx1250) — matrix operand order repaired and checked by reading, not compiled, not run on hardware
//
#include <hip/hip_runtime.h>

typedef __attribute__((ext_vector_type(16))) __bf16 v16bf;
typedef __attribute__((ext_vector_type(8)))  float  v8f;
typedef __attribute__((ext_vector_type(4)))  int    v4i;

#define B_    16
#define CIN   512
#define COUT  512
#define H_    32
#define W_    32
#define KSZ   3
#define WDIM  512
#define HW    (H_ * W_)
#define KTOT  (KSZ * KSZ * CIN)     // 4608
#define NKSTEP (KTOT / 32)          // 144
#define HP    (H_ + 2)              // 34 padded
#define WP    (W_ + 2)              // 34 padded

#if defined(__has_builtin)
#if __has_builtin(__builtin_amdgcn_global_load_async_to_lds_b128) && \
    __has_builtin(__builtin_amdgcn_s_wait_asynccnt)
#define USE_ASYNC_LDS 1
#endif
#endif
#ifndef USE_ASYNC_LDS
#define USE_ASYNC_LDS 0
#endif

union Frag { uint4 u[2]; v16bf v; };

#if USE_ASYNC_LDS
typedef __attribute__((address_space(1))) v4i* as1_v4i;
typedef __attribute__((address_space(3))) v4i* as3_v4i;
__device__ __forceinline__ void async_copy_b128(const unsigned short* g, unsigned short* l) {
    __builtin_amdgcn_global_load_async_to_lds_b128((as1_v4i)g, (as3_v4i)l, 0, 0);
}
#endif

// round-to-nearest-even float -> bf16 bits
__device__ __forceinline__ unsigned short f2bf(float f) {
    unsigned int u = __float_as_uint(f);
    unsigned int r = (u + 0x7FFFu + ((u >> 16) & 1u)) >> 16;
    return (unsigned short)r;
}

// ---------------- style = w_latent @ dense_kernel + bias : [B, CIN] ----------------
__global__ __launch_bounds__(256) void style_kernel(
    const float* __restrict__ wl, const float* __restrict__ dk,
    const float* __restrict__ bias, float* __restrict__ style) {
    const int c = blockIdx.x * 256 + threadIdx.x;
    const int b = blockIdx.y;
    float acc = bias[c];
    for (int w = 0; w < WDIM; ++w)
        acc = fmaf(wl[b * WDIM + w], dk[w * CIN + c], acc);
    style[b * CIN + c] = acc;
}

// ---------------- T[kh][ci] = sum_kw sum_co conv_w^2 : [3, CIN] ----------------
__global__ __launch_bounds__(256) void tnorm_kernel(
    const float* __restrict__ cw, float* __restrict__ T) {
    const int kh = blockIdx.x / CIN;
    const int ci = blockIdx.x % CIN;
    float acc = 0.f;
    for (int i = threadIdx.x; i < KSZ * COUT; i += 256) {
        const int kw = i / COUT, co = i % COUT;
        const float v = cw[(((kh * KSZ) + kw) * CIN + ci) * COUT + co];
        acc = fmaf(v, v, acc);
    }
    __shared__ float red[256];
    red[threadIdx.x] = acc;
    __syncthreads();
    for (int s = 128; s > 0; s >>= 1) {
        if (threadIdx.x < s) red[threadIdx.x] += red[threadIdx.x + s];
        __syncthreads();
    }
    if (threadIdx.x == 0) T[blockIdx.x] = red[0];
}

// ---------------- denom[b][kh] = rsqrt(sum_ci style^2 * T) : [B, 3] ----------------
__global__ __launch_bounds__(256) void denom_kernel(
    const float* __restrict__ style, const float* __restrict__ T,
    float* __restrict__ denom) {
    const int b = blockIdx.x / KSZ, kh = blockIdx.x % KSZ;
    float acc = 0.f;
    for (int ci = threadIdx.x; ci < CIN; ci += 256) {
        const float s = style[b * CIN + ci];
        acc = fmaf(s * s, T[kh * CIN + ci], acc);
    }
    __shared__ float red[256];
    red[threadIdx.x] = acc;
    __syncthreads();
    for (int s = 128; s > 0; s >>= 1) {
        if (threadIdx.x < s) red[threadIdx.x] += red[threadIdx.x + s];
        __syncthreads();
    }
    if (threadIdx.x == 0) denom[blockIdx.x] = rsqrtf(red[0]);
}

// ---------------- scale[b][kh][ci] = style[b][ci] * denom[b][kh] ----------------
__global__ __launch_bounds__(256) void scale_kernel(
    const float* __restrict__ style, const float* __restrict__ denom,
    float* __restrict__ scale) {
    const int i = blockIdx.x * 256 + threadIdx.x;
    const int b  = i / (KSZ * CIN);
    const int kh = (i / CIN) % KSZ;
    const int ci = i % CIN;
    scale[i] = style[b * CIN + ci] * denom[b * KSZ + kh];
}

// ---------------- zero the padded bf16 image buffer (uint4 fill) ----------------
__global__ __launch_bounds__(256) void zero_imgp_kernel(uint4* __restrict__ p) {
    const uint4 z = {0u, 0u, 0u, 0u};
    p[blockIdx.x * 256 + threadIdx.x] = z;
}

// ---------------- NCHW fp32 image -> padded NHWC bf16: imgp[b][y+1][x+1][ci] ----------------
__global__ __launch_bounds__(256) void img_pack_kernel(
    const float* __restrict__ image, unsigned short* __restrict__ imgp) {
    __shared__ float tile[32][33];
    const int y = blockIdx.x;
    const int b = blockIdx.y;
    const int t = threadIdx.x;
    for (int cb = 0; cb < CIN; cb += 32) {
        {   // load 32ci x 32x, coalesced along x
            const int x = t & 31, cr4 = (t >> 5) * 4;
#pragma unroll
            for (int u = 0; u < 4; ++u) {
                const int ci = cb + cr4 + u;
                tile[cr4 + u][x] = image[(((size_t)b * CIN + ci) * H_ + y) * W_ + x];
            }
        }
        __syncthreads();
        {   // store: pack 4 bf16 along ci, coalesced along ci
            const int x = t >> 3, cl4 = (t & 7) * 4;
            const unsigned int lo = (unsigned)f2bf(tile[cl4 + 0][x]) |
                                    ((unsigned)f2bf(tile[cl4 + 1][x]) << 16);
            const unsigned int hi = (unsigned)f2bf(tile[cl4 + 2][x]) |
                                    ((unsigned)f2bf(tile[cl4 + 3][x]) << 16);
            uint2 pk; pk.x = lo; pk.y = hi;
            *(uint2*)&imgp[(((size_t)b * HP + (y + 1)) * WP + (x + 1)) * CIN + cb + cl4] = pk;
        }
        __syncthreads();
    }
}

// ---------------- modulated bf16 weights, transposed: wbf[b][co][k], k = khkw*512+ci ----------------
__global__ __launch_bounds__(256) void wmod_kernel(
    const float* __restrict__ cw, const float* __restrict__ scale,
    unsigned short* __restrict__ wbf) {
    __shared__ float tile[32][33];  // [krel][corel]
    const int kB  = blockIdx.x * 32;
    const int coB = blockIdx.y * 32;
    const int b   = blockIdx.z;
    const int t   = threadIdx.x;
    {   // read coalesced along co
        const int co = coB + (t & 31), kr4 = (t >> 5) * 4;
#pragma unroll
        for (int u = 0; u < 4; ++u) {
            const int k  = kB + kr4 + u;
            const int ci = k & (CIN - 1);
            const int kh = (k >> 9) / 3;
            tile[kr4 + u][t & 31] =
                cw[(size_t)k * COUT + co] * scale[((size_t)b * KSZ + kh) * CIN + ci];
        }
    }
    __syncthreads();
    {   // write coalesced along k
        const int krel = t & 31, c4 = (t >> 5) * 4;
        const int k = kB + krel;
#pragma unroll
        for (int u = 0; u < 4; ++u) {
            const int co = coB + c4 + u;
            wbf[((size_t)b * COUT + co) * KTOT + k] = f2bf(tile[krel][c4 + u]);
        }
    }
}

// ---------------- main WMMA GEMM: out[b,co,p] = sum_k wbf[b,co,k] * patch[b,k,p] ----------------
// Block tile: 128 cout (M) x 128 pixels (N), BK = 32. 8 waves as 4(M) x 2(N);
// each wave: 2 A-frags x 4 B-frags -> 8 x v_wmma_f32_16x16x32_bf16 per K-step.
// Staging: GLOBAL_LOAD_ASYNC_TO_LDS_B128 + LDS ping-pong (ASYNCcnt), fallback =
// register double-buffering if the async builtins are unavailable.
__global__ __launch_bounds__(256) void modconv_wmma_kernel(
    const unsigned short* __restrict__ wbf,
    const unsigned short* __restrict__ imgp,
    float* __restrict__ out) {
#if USE_ASYNC_LDS
    __shared__ __attribute__((aligned(16))) unsigned short Ws[2][128 * 40];
    __shared__ __attribute__((aligned(16))) unsigned short Im[2][128 * 40];
#else
    __shared__ __attribute__((aligned(16))) unsigned short Ws[1][128 * 40];
    __shared__ __attribute__((aligned(16))) unsigned short Im[1][128 * 40];
#endif

    const int tid     = threadIdx.x;
    const int pixBase = blockIdx.x * 128;
    const int coBase  = blockIdx.y * 128;
    const int b       = blockIdx.z;

    const int lane = tid & 31, wave = tid >> 5;
    const int mw = wave >> 1;   // 0..3 : 32-cout sub-block
    const int nw = wave & 1;    // 0..1 : 64-pixel sub-block
    const int g  = lane >> 4, r = lane & 15;

    // staging geometry: this thread owns rows {row0, row0+64}, 16B chunk q
    const int row0 = tid >> 2;
    const int q    = tid & 3;

    const unsigned short* aSrc0 = wbf + ((size_t)b * COUT + coBase + row0) * KTOT + q * 8;
    const unsigned short* aSrc1 = aSrc0 + (size_t)64 * KTOT;
    const int p0 = pixBase + row0;
    const int y0 = p0 >> 5, x0 = p0 & 31;
    const unsigned short* bSrc0 = imgp + (((size_t)b * HP + y0) * WP + x0) * CIN + q * 8;
    const unsigned short* bSrc1 = bSrc0 + (size_t)2 * WP * CIN;  // +64 pixels = +2 rows

    const int dOff0 = row0 * 40 + q * 8;
    const int dOff1 = (row0 + 64) * 40 + q * 8;
    const int aOff0 = (mw * 32 + r) * 40;
    const int aOff1 = (mw * 32 + 16 + r) * 40;
    const int bOff  = (nw * 64 + r) * 40 + g * 8;

    v8f acc[2][4] = {};

#if USE_ASYNC_LDS
    // ---- async-staged, LDS ping-pong ----
    {
        // step 0 into buffer 0 (khkw=0 -> B offset 0)
        async_copy_b128(aSrc0, &Ws[0][dOff0]);
        async_copy_b128(aSrc1, &Ws[0][dOff1]);
        async_copy_b128(bSrc0, &Im[0][dOff0]);
        async_copy_b128(bSrc1, &Im[0][dOff1]);
    }
    for (int s = 0; s < NKSTEP; ++s) {
        const int cur = s & 1;
        if (s + 1 < NKSTEP) {
            const int s1   = s + 1;
            const int khkw = s1 >> 4;
            const int kh   = khkw / 3;
            const int kw   = khkw - 3 * kh;
            const int offB = ((kh * WP + kw) << 9) + ((s1 & 15) << 5); // halves
            const int nxt  = cur ^ 1;
            async_copy_b128(aSrc0 + (size_t)s1 * 32, &Ws[nxt][dOff0]);
            async_copy_b128(aSrc1 + (size_t)s1 * 32, &Ws[nxt][dOff1]);
            async_copy_b128(bSrc0 + offB,            &Im[nxt][dOff0]);
            async_copy_b128(bSrc1 + offB,            &Im[nxt][dOff1]);
            __builtin_amdgcn_s_wait_asynccnt(4);  // step-s group complete (in-order)
        } else {
            __builtin_amdgcn_s_wait_asynccnt(0);
        }
        __syncthreads();
        __builtin_prefetch(aSrc0 + (size_t)(s + 8) * 32, 0, 1);

        Frag a0, a1, bq[4];
        const unsigned short* wsb = &Ws[cur][0];
        const unsigned short* imb = &Im[cur][0];
        a0.u[0] = *(const uint4*)(wsb + aOff0 + g * 8);
        a0.u[1] = *(const uint4*)(wsb + aOff0 + 16 + g * 8);
        a1.u[0] = *(const uint4*)(wsb + aOff1 + g * 8);
        a1.u[1] = *(const uint4*)(wsb + aOff1 + 16 + g * 8);
#pragma unroll
        for (int j = 0; j < 4; ++j) {
            bq[j].u[0] = *(const uint4*)(imb + bOff + j * (16 * 40));
            bq[j].u[1] = *(const uint4*)(imb + bOff + j * (16 * 40) + 16);
        }
#pragma unroll
        for (int j = 0; j < 4; ++j) {
            acc[0][j] = __builtin_amdgcn_wmma_f32_16x16x32_bf16(
                false, a0.v, false, bq[j].v, (short)0, acc[0][j], false, false);
            acc[1][j] = __builtin_amdgcn_wmma_f32_16x16x32_bf16(
                false, a1.v, false, bq[j].v, (short)0, acc[1][j], false, false);
        }
        __syncthreads();
    }
#else
    // ---- fallback: register double-buffering into a single LDS buffer ----
    uint4* wDst0 = (uint4*)&Ws[0][dOff0];
    uint4* wDst1 = (uint4*)&Ws[0][dOff1];
    uint4* iDst0 = (uint4*)&Im[0][dOff0];
    uint4* iDst1 = (uint4*)&Im[0][dOff1];

    uint4 ra0 = *(const uint4*)(aSrc0);
    uint4 ra1 = *(const uint4*)(aSrc1);
    uint4 rb0 = *(const uint4*)(bSrc0);
    uint4 rb1 = *(const uint4*)(bSrc1);

    for (int s = 0; s < NKSTEP; ++s) {
        *wDst0 = ra0; *wDst1 = ra1;
        *iDst0 = rb0; *iDst1 = rb1;
        __syncthreads();

        uint4 na0 = {0,0,0,0}, na1 = {0,0,0,0}, nb0 = {0,0,0,0}, nb1 = {0,0,0,0};
        if (s + 1 < NKSTEP) {
            const int s1   = s + 1;
            const int khkw = s1 >> 4;
            const int kh   = khkw / 3;
            const int kw   = khkw - 3 * kh;
            const int offB = ((kh * WP + kw) << 9) + ((s1 & 15) << 5);
            na0 = *(const uint4*)(aSrc0 + (size_t)s1 * 32);
            na1 = *(const uint4*)(aSrc1 + (size_t)s1 * 32);
            nb0 = *(const uint4*)(bSrc0 + offB);
            nb1 = *(const uint4*)(bSrc1 + offB);
        }
        __builtin_prefetch(aSrc0 + (size_t)(s + 8) * 32, 0, 1);

        Frag a0, a1, bq[4];
        a0.u[0] = *(const uint4*)(&Ws[0][aOff0] + g * 8);
        a0.u[1] = *(const uint4*)(&Ws[0][aOff0] + 16 + g * 8);
        a1.u[0] = *(const uint4*)(&Ws[0][aOff1] + g * 8);
        a1.u[1] = *(const uint4*)(&Ws[0][aOff1] + 16 + g * 8);
#pragma unroll
        for (int j = 0; j < 4; ++j) {
            bq[j].u[0] = *(const uint4*)(&Im[0][bOff + j * (16 * 40)]);
            bq[j].u[1] = *(const uint4*)(&Im[0][bOff + j * (16 * 40)] + 16);
        }
#pragma unroll
        for (int j = 0; j < 4; ++j) {
            acc[0][j] = __builtin_amdgcn_wmma_f32_16x16x32_bf16(
                false, a0.v, false, bq[j].v, (short)0, acc[0][j], false, false);
            acc[1][j] = __builtin_amdgcn_wmma_f32_16x16x32_bf16(
                false, a1.v, false, bq[j].v, (short)0, acc[1][j], false, false);
        }
        __syncthreads();

        ra0 = na0; ra1 = na1; rb0 = nb0; rb1 = nb1;
    }
#endif

    // ---- epilogue: D elem e -> co = +e+8g ; pixel = r  => coalesced 64B stores ----
#pragma unroll
    for (int i = 0; i < 2; ++i) {
        const int coRow = coBase + mw * 32 + i * 16 + 8 * g;
#pragma unroll
        for (int e = 0; e < 8; ++e) {
            const size_t o = ((size_t)b * COUT + (coRow + e)) * (size_t)HW;
#pragma unroll
            for (int j = 0; j < 4; ++j)
                out[o + pixBase + nw * 64 + j * 16 + r] = acc[i][j][e];
        }
    }
}

extern "C" void kernel_launch(void* const* d_in, const int* in_sizes, int n_in,
                              void* d_out, int out_size, void* d_ws, size_t ws_size,
                              hipStream_t stream) {
    const float* image = (const float*)d_in[0];   // [16,512,32,32]
    const float* wl    = (const float*)d_in[1];   // [16,512]
    const float* dk    = (const float*)d_in[2];   // [512,512]
    const float* bias  = (const float*)d_in[3];   // [512]
    const float* cw    = (const float*)d_in[4];   // [3,3,512,512]
    float* out = (float*)d_out;                   // [16,512,32,32]

    // workspace layout (~94.6 MB total, fits global L2):
    unsigned short* wbf  = (unsigned short*)d_ws;            // 16*512*4608  = 75.5 MB
    unsigned short* imgp = wbf + (size_t)B_ * COUT * KTOT;   // 16*34*34*512 = 18.9 MB
    float* style = (float*)(imgp + (size_t)B_ * HP * WP * CIN);
    float* T     = style + B_ * CIN;
    float* denom = T + KSZ * CIN;
    float* scl   = denom + B_ * KSZ;                         // 16*3*512

    style_kernel<<<dim3(CIN / 256, B_), 256, 0, stream>>>(wl, dk, bias, style);
    tnorm_kernel<<<KSZ * CIN, 256, 0, stream>>>(cw, T);
    denom_kernel<<<B_ * KSZ, 256, 0, stream>>>(style, T, denom);
    scale_kernel<<<(B_ * KSZ * CIN) / 256, 256, 0, stream>>>(style, denom, scl);

    const int imgpVec = (B_ * HP * WP * CIN) / 8;            // uint4 count
    zero_imgp_kernel<<<imgpVec / 256, 256, 0, stream>>>((uint4*)imgp);
    img_pack_kernel<<<dim3(H_, B_), 256, 0, stream>>>(image, imgp);
    wmod_kernel<<<dim3(KTOT / 32, COUT / 32, B_), 256, 0, stream>>>(cw, scl, wbf);

    modconv_wmma_kernel<<<dim3(HW / 128, COUT / 128, B_), 256, 0, stream>>>(wbf, imgp, out);
}
